// AttentionCopyDecoder_35373350650618
// MI455X (gfx1250) — matrix operand order repaired and checked by reading, not compiled, not run on hardware
//
#include <hip/hip_runtime.h>

// ---------------- shapes ----------------
#define BB   64
#define TT   400
#define H2   1024
#define EE   512
#define VV   50000
#define K_OUT 2560   // E + 2*H2

typedef __attribute__((ext_vector_type(16))) __bf16 v16bf;
typedef __attribute__((ext_vector_type(8)))  float  v8f;

union BF16x16 { v16bf v; unsigned int u[8]; };

// Pack two f32 into two bf16 (truncating) with a single v_perm_b32:
// dst[15:0] = a[31:16], dst[31:16] = b[31:16]
__device__ __forceinline__ unsigned int pack2bf(float a, float b) {
    return __builtin_amdgcn_perm(__float_as_uint(b), __float_as_uint(a), 0x07060302u);
}
__device__ __forceinline__ float sigf(float x) { return 1.0f / (1.0f + __expf(-x)); }

// A-matrix k offset for vgpr v, lane-half hi (16-bit A 16x32 layout, ISA 7.12.2)
__device__ __forceinline__ int ka_off(int v, int hi) {
    return ((v & 4) ? 16 : 0) + hi * 8 + (v & 3) * 2;
}
// B-matrix k offset (same k order as A: lanes 0-15 K=0..7,16..23, lanes 16-31 K=8..15,24..31)
__device__ __forceinline__ int kb_off(int v, int hi) {
    return ((v & 4) ? 16 : 0) + hi * 8 + (v & 3) * 2;
}

// ---------------------------------------------------------------------------
// 1) alignment logits: aw_raw[b,t] = tanh(enc[b,t]·W[0:H2] + ph[b]·W[H2:2H2] + b)
// ---------------------------------------------------------------------------
__global__ void aw_logit_kernel(const float* __restrict__ enc_out,
                                const float* __restrict__ pre_hidden,
                                const float* __restrict__ align_W,
                                const float* __restrict__ align_b,
                                float* __restrict__ aw_raw) {
    int bt = blockIdx.x;                 // b*TT + t
    int b  = bt / TT;
    int tid = threadIdx.x;               // 128
    __shared__ float red[128];
    float s = 0.f;
    for (int k = tid; k < H2; k += 128) {
        s += enc_out[(size_t)bt * H2 + k]   * align_W[k];
        s += pre_hidden[(size_t)b * H2 + k] * align_W[H2 + k];
    }
    red[tid] = s; __syncthreads();
    for (int st = 64; st > 0; st >>= 1) {
        if (tid < st) red[tid] += red[tid + st];
        __syncthreads();
    }
    if (tid == 0) aw_raw[bt] = tanhf(red[0] + align_b[0]);
}

// ---------------------------------------------------------------------------
// 2) row softmax (block per row), optional second destination
// ---------------------------------------------------------------------------
__global__ void softmax_rows_kernel(const float* __restrict__ src,
                                    float* __restrict__ dst,
                                    float* __restrict__ dst2, int L) {
    int row = blockIdx.x;
    int tid = threadIdx.x;               // 256
    __shared__ float red[256];
    const float* s = src + (size_t)row * L;
    float mx = -3.4e38f;
    for (int i = tid; i < L; i += 256) mx = fmaxf(mx, s[i]);
    red[tid] = mx; __syncthreads();
    for (int st = 128; st > 0; st >>= 1) {
        if (tid < st) red[tid] = fmaxf(red[tid], red[tid + st]);
        __syncthreads();
    }
    mx = red[0]; __syncthreads();
    float sum = 0.f;
    for (int i = tid; i < L; i += 256) sum += __expf(s[i] - mx);
    red[tid] = sum; __syncthreads();
    for (int st = 128; st > 0; st >>= 1) {
        if (tid < st) red[tid] += red[tid + st];
        __syncthreads();
    }
    float inv = 1.0f / red[0];
    for (int i = tid; i < L; i += 256) {
        float v = __expf(s[i] - mx) * inv;
        dst[(size_t)row * L + i] = v;
        if (dst2) dst2[(size_t)row * L + i] = v;
    }
}

// ---------------------------------------------------------------------------
// 3) context + readM (block per b)
// ---------------------------------------------------------------------------
__global__ void ctx_readm_kernel(const float* __restrict__ aw_bt,
                                 const float* __restrict__ enc_out,
                                 const int* __restrict__ source_input,
                                 const int* __restrict__ input_ids,
                                 float* __restrict__ context,
                                 float* __restrict__ readM) {
    int b = blockIdx.x;
    int tid = threadIdx.x;               // 256
    __shared__ float aw_s[TT];
    __shared__ int   m_s[TT];
    __shared__ float red[256];
    __shared__ float inv_s;
    int tgt = input_ids[b];
    for (int t = tid; t < TT; t += 256) {
        aw_s[t] = aw_bt[b * TT + t];
        m_s[t]  = (source_input[b * TT + t] == tgt);
    }
    __syncthreads();
    float wv = 0.f;
    for (int t = tid; t < TT; t += 256) wv += m_s[t] ? aw_s[t] : 0.f;
    red[tid] = wv; __syncthreads();
    for (int st = 128; st > 0; st >>= 1) {
        if (tid < st) red[tid] += red[tid + st];
        __syncthreads();
    }
    if (tid == 0) inv_s = 1.0f / (red[0] + 1e-8f);
    __syncthreads();
    float inv = inv_s;
    for (int c = tid; c < H2; c += 256) {
        float ctx = 0.f, rm = 0.f;
        const float* e = enc_out + (size_t)b * TT * H2 + c;
        for (int t = 0; t < TT; ++t) {
            float ev = e[(size_t)t * H2];
            float a  = aw_s[t];
            ctx += a * ev;
            if (m_s[t]) rm += a * ev;
        }
        context[b * H2 + c] = ctx;
        readM[b * H2 + c]   = rm * inv;
    }
}

// ---------------------------------------------------------------------------
// 4) concat + f32->bf16 pack (up to 3 sources, optional gather on src0, relu)
// ---------------------------------------------------------------------------
__device__ __forceinline__ float pk_fetch(int b, int c,
        const float* s0, int l0, const int* idx0,
        const float* s1, int l1, const float* s2, int l2) {
    if (c < l0) {
        int r = idx0 ? idx0[b] : b;
        return s0[(size_t)r * l0 + c];
    }
    c -= l0;
    if (c < l1) return s1[(size_t)b * l1 + c];
    return s2[(size_t)b * l2 + (c - l1)];
}

__global__ void pack_bf16_kernel(unsigned int* __restrict__ dst,
                                 const float* s0, int l0, const int* idx0,
                                 const float* s1, int l1,
                                 const float* s2, int l2, int relu) {
    int Ktot = l0 + l1 + l2;
    int rowlen = Ktot >> 1;
    int total = BB * rowlen;
    for (int i = blockIdx.x * blockDim.x + threadIdx.x; i < total;
         i += gridDim.x * blockDim.x) {
        int b = i / rowlen;
        int c = (i % rowlen) * 2;
        float f0 = pk_fetch(b, c,     s0, l0, idx0, s1, l1, s2, l2);
        float f1 = pk_fetch(b, c + 1, s0, l0, idx0, s1, l1, s2, l2);
        if (relu) { f0 = fmaxf(f0, 0.f); f1 = fmaxf(f1, 0.f); }
        dst[i] = pack2bf(f0, f1);
    }
}

__global__ void cvt_bf16_kernel(unsigned int* __restrict__ dst,
                                const float* __restrict__ src, int npairs) {
    int i = blockIdx.x * blockDim.x + threadIdx.x;
    if (i < npairs) {
        float2 v = *(const float2*)(src + 2 * (size_t)i);
        dst[i] = pack2bf(v.x, v.y);
    }
}

// ---------------------------------------------------------------------------
// 5) Generic WMMA GEMM:  out[64,N] = A[64,K](bf16) @ W[N,K](f32->bf16)^T + bias
//    One wave per 16-wide N tile, 4 M tiles (M=64), K in chunks of 32.
// ---------------------------------------------------------------------------
__global__ __launch_bounds__(32) void gemm_a64_wmma(
        const unsigned int* __restrict__ A,   // bf16 pairs: 64 x (K/2)
        const float* __restrict__ W,          // N x K
        const float* __restrict__ bias,       // N (nullable)
        float* __restrict__ out,              // 64 x N
        int N, int K) {
    int n0   = blockIdx.x * 16;
    int lane = threadIdx.x;
    int col  = lane & 15;
    int hi   = lane >> 4;
    int Kp   = K >> 1;
    v8f acc0 = {}, acc1 = {}, acc2 = {}, acc3 = {};
    const float* wrow = W + (size_t)(n0 + col) * K;
    const unsigned int* a0p = A + (size_t)(col)      * Kp;
    const unsigned int* a1p = A + (size_t)(16 + col) * Kp;
    const unsigned int* a2p = A + (size_t)(32 + col) * Kp;
    const unsigned int* a3p = A + (size_t)(48 + col) * Kp;
    for (int kc = 0; kc < K; kc += 32) {
        BF16x16 bm, a0, a1, a2, a3;
#pragma unroll
        for (int v = 0; v < 8; ++v) {
            int kb = kc + kb_off(v, hi);
            float2 w2 = *(const float2*)(wrow + kb);
            bm.u[v] = pack2bf(w2.x, w2.y);
            int o = (kc + ka_off(v, hi)) >> 1;
            a0.u[v] = a0p[o]; a1.u[v] = a1p[o];
            a2.u[v] = a2p[o]; a3.u[v] = a3p[o];
        }
        acc0 = __builtin_amdgcn_wmma_f32_16x16x32_bf16(false, a0.v, false, bm.v, (short)0, acc0, false, false);
        acc1 = __builtin_amdgcn_wmma_f32_16x16x32_bf16(false, a1.v, false, bm.v, (short)0, acc1, false, false);
        acc2 = __builtin_amdgcn_wmma_f32_16x16x32_bf16(false, a2.v, false, bm.v, (short)0, acc2, false, false);
        acc3 = __builtin_amdgcn_wmma_f32_16x16x32_bf16(false, a3.v, false, bm.v, (short)0, acc3, false, false);
    }
    float bv = bias ? bias[n0 + col] : 0.0f;
#pragma unroll
    for (int r = 0; r < 8; ++r) {
        int m = r + hi * 8;
        out[(size_t)(m)      * N + n0 + col] = acc0[r] + bv;
        out[(size_t)(m + 16) * N + n0 + col] = acc1[r] + bv;
        out[(size_t)(m + 32) * N + n0 + col] = acc2[r] + bv;
        out[(size_t)(m + 48) * N + n0 + col] = acc3[r] + bv;
    }
}

// ---------------------------------------------------------------------------
// 6) GRU gates
// ---------------------------------------------------------------------------
__global__ void gru_gate_kernel(const float* __restrict__ gi,
                                const float* __restrict__ gh,
                                const float* __restrict__ pre_hidden,
                                float* __restrict__ h_new,
                                float* __restrict__ cur_h_out) {
    int i = blockIdx.x * blockDim.x + threadIdx.x;
    if (i >= BB * H2) return;
    int b = i >> 10, c = i & (H2 - 1);
    size_t o = (size_t)b * 3 * H2;
    float r  = sigf(gi[o + c]          + gh[o + c]);
    float z  = sigf(gi[o + H2 + c]     + gh[o + H2 + c]);
    float n  = tanhf(gi[o + 2 * H2 + c] + r * gh[o + 2 * H2 + c]);
    float h  = pre_hidden[i];
    float hn = (1.f - z) * n + z * h;
    h_new[i] = hn;
    cur_h_out[i] = hn;
}

// ---------------------------------------------------------------------------
// 7) fused ca/caw: caw_logit[b,t] = sum_d sigmoid(enc[b,t]·copyW[d] + cb[d]) * h_new[b,d]
//    One wave per 16-row tile of (B*T); A tile staged in LDS as bf16.
// ---------------------------------------------------------------------------
__global__ __launch_bounds__(32) void ca_caw_kernel(
        const float* __restrict__ enc_out,        // (B*T, H2)
        const unsigned int* __restrict__ copyWb,  // bf16 pairs: H2 x (H2/2)
        const float* __restrict__ copy_b,
        const float* __restrict__ h_new,          // B x H2
        float* __restrict__ caw_logit) {          // B*T
    __shared__ unsigned int As[16 * (H2 / 2)];    // 32 KB
    __shared__ float Ps[32][8];
    int r0   = blockIdx.x * 16;                   // row tile (stays within one b: T%16==0)
    int b    = r0 / TT;
    int lane = threadIdx.x;
    for (int i = lane; i < 16 * (H2 / 2); i += 32) {
        int m = i >> 9;                           // /(H2/2)
        int c = i & (H2 / 2 - 1);
        float2 e2 = *(const float2*)(enc_out + (size_t)(r0 + m) * H2 + c * 2);
        As[i] = pack2bf(e2.x, e2.y);
    }
    __syncthreads();
    int col = lane & 15, hi = lane >> 4;
    float partial[8];
#pragma unroll
    for (int r = 0; r < 8; ++r) partial[r] = 0.f;
    const float* hrow = h_new + (size_t)b * H2;
    for (int d0 = 0; d0 < H2; d0 += 16) {
        v8f acc = {};
        const unsigned int* wrow = copyWb + (size_t)(d0 + col) * (H2 / 2);
        for (int kc = 0; kc < H2; kc += 32) {
            BF16x16 am, bm;
#pragma unroll
            for (int v = 0; v < 8; ++v) {
                am.u[v] = As[col * (H2 / 2) + ((kc + ka_off(v, hi)) >> 1)];
                bm.u[v] = wrow[(kc + kb_off(v, hi)) >> 1];
            }
            acc = __builtin_amdgcn_wmma_f32_16x16x32_bf16(false, am.v, false, bm.v, (short)0, acc, false, false);
        }
        int d = d0 + col;
        float cb = copy_b[d];
        float hv = hrow[d];
#pragma unroll
        for (int r = 0; r < 8; ++r)
            partial[r] += sigf(acc[r] + cb) * hv;
    }
#pragma unroll
    for (int r = 0; r < 8; ++r) Ps[lane][r] = partial[r];
    __syncthreads();
    if (lane < 16) {                              // row m: half h = m>=8, vgpr r = m&7
        int m = lane, h = (m < 8) ? 0 : 1, r = m & 7;
        float s = 0.f;
        for (int l = 0; l < 16; ++l) s += Ps[h * 16 + l][r];
        caw_logit[r0 + m] = s;
    }
}

// ---------------------------------------------------------------------------
// 8) deterministic scatter-add of copy probabilities into d_out (one thread/row)
// ---------------------------------------------------------------------------
__global__ void spp_kernel(const int* __restrict__ source_input,
                           const float* __restrict__ aw_bt,
                           float* __restrict__ out) {   // B x V
    int b = blockIdx.x * blockDim.x + threadIdx.x;
    if (b >= BB) return;
    float* row = out + (size_t)b * VV;
    for (int t = 0; t < TT; ++t)
        row[source_input[b * TT + t]] += aw_bt[b * TT + t];
}

// ---------------------------------------------------------------------------
extern "C" void kernel_launch(void* const* d_in, const int* in_sizes, int n_in,
                              void* d_out, int out_size, void* d_ws, size_t ws_size,
                              hipStream_t stream) {
    const int*   input_ids    = (const int*)  d_in[0];
    const float* pre_hidden   = (const float*)d_in[1];
    const float* enc_out      = (const float*)d_in[2];
    const int*   source_input = (const int*)  d_in[3];
    const float* emb_table    = (const float*)d_in[4];
    const float* align_W      = (const float*)d_in[5];
    const float* align_b      = (const float*)d_in[6];
    const float* dnn_W        = (const float*)d_in[7];
    const float* dnn_b        = (const float*)d_in[8];
    const float* gru_Wih      = (const float*)d_in[9];
    const float* gru_Whh      = (const float*)d_in[10];
    const float* gru_bih      = (const float*)d_in[11];
    const float* gru_bhh      = (const float*)d_in[12];
    const float* comb_W       = (const float*)d_in[13];
    const float* comb_b       = (const float*)d_in[14];
    const float* out_W        = (const float*)d_in[15];
    const float* out_b        = (const float*)d_in[16];
    const float* copy_W       = (const float*)d_in[17];
    const float* copy_b       = (const float*)d_in[18];

    float* o_out  = (float*)d_out;               // B*V
    float* o_curh = o_out  + (size_t)BB * VV;    // B*H2
    float* o_aw   = o_curh + (size_t)BB * H2;    // B*T
    float* o_caw  = o_aw   + (size_t)BB * TT;    // B*T

    char* wp = (char*)d_ws;
    auto alloc = [&](size_t bytes) -> void* {
        void* p = wp; wp += (bytes + 255) & ~(size_t)255; return p;
    };
    float* aw_raw        = (float*)alloc((size_t)BB * TT * 4);
    float* aw_bt         = (float*)alloc((size_t)BB * TT * 4);
    float* context       = (float*)alloc((size_t)BB * H2 * 4);
    float* readM         = (float*)alloc((size_t)BB * H2 * 4);
    unsigned int* u1     = (unsigned int*)alloc((size_t)BB * 768  * 4); // [emb,readM] K=1536
    float* emb2          = (float*)alloc((size_t)BB * EE * 4);
    unsigned int* u2     = (unsigned int*)alloc((size_t)BB * 768  * 4); // [emb2,context]
    float* comb_out      = (float*)alloc((size_t)BB * EE * 4);
    unsigned int* xbf    = (unsigned int*)alloc((size_t)BB * 256  * 4); // relu(comb) K=512
    unsigned int* hbf    = (unsigned int*)alloc((size_t)BB * 512  * 4); // pre_hidden K=1024
    float* gi            = (float*)alloc((size_t)BB * 3 * H2 * 4);
    float* gh            = (float*)alloc((size_t)BB * 3 * H2 * 4);
    float* h_new         = (float*)alloc((size_t)BB * H2 * 4);
    unsigned int* uo     = (unsigned int*)alloc((size_t)BB * 1280 * 4); // K=2560
    unsigned int* copyWb = (unsigned int*)alloc((size_t)H2 * (H2 / 2) * 4);
    float* logits        = (float*)alloc((size_t)BB * VV * 4);
    float* caw_log       = (float*)alloc((size_t)BB * TT * 4);

    // attention weights
    aw_logit_kernel<<<BB * TT, 128, 0, stream>>>(enc_out, pre_hidden, align_W, align_b, aw_raw);
    softmax_rows_kernel<<<BB, 256, 0, stream>>>(aw_raw, aw_bt, o_aw, TT);
    ctx_readm_kernel<<<BB, 256, 0, stream>>>(aw_bt, enc_out, source_input, input_ids, context, readM);

    // emb2 = [emb, readM] @ dnn_W^T + dnn_b
    pack_bf16_kernel<<<192, 256, 0, stream>>>(u1, emb_table, EE, input_ids, readM, H2, nullptr, 0, 0);
    gemm_a64_wmma<<<EE / 16, 32, 0, stream>>>(u1, dnn_W, dnn_b, emb2, EE, EE + H2);

    // x = relu([emb2, context] @ comb_W^T + comb_b)
    pack_bf16_kernel<<<192, 256, 0, stream>>>(u2, emb2, EE, nullptr, context, H2, nullptr, 0, 0);
    gemm_a64_wmma<<<EE / 16, 32, 0, stream>>>(u2, comb_W, comb_b, comb_out, EE, EE + H2);
    pack_bf16_kernel<<<64, 256, 0, stream>>>(xbf, comb_out, EE, nullptr, nullptr, 0, nullptr, 0, 1);
    pack_bf16_kernel<<<128, 256, 0, stream>>>(hbf, pre_hidden, H2, nullptr, nullptr, 0, nullptr, 0, 0);

    // GRU
    gemm_a64_wmma<<<3 * H2 / 16, 32, 0, stream>>>(xbf, gru_Wih, gru_bih, gi, 3 * H2, EE);
    gemm_a64_wmma<<<3 * H2 / 16, 32, 0, stream>>>(hbf, gru_Whh, gru_bhh, gh, 3 * H2, H2);
    gru_gate_kernel<<<(BB * H2 + 255) / 256, 256, 0, stream>>>(gi, gh, pre_hidden, h_new, o_curh);

    // vocab logits + softmax + copy-prob scatter
    pack_bf16_kernel<<<320, 256, 0, stream>>>(uo, emb2, EE, nullptr, context, H2, h_new, H2, 0);
    gemm_a64_wmma<<<VV / 16, 32, 0, stream>>>(uo, out_W, out_b, logits, VV, K_OUT);
    softmax_rows_kernel<<<BB, 256, 0, stream>>>(logits, o_out, nullptr, VV);
    spp_kernel<<<2, 32, 0, stream>>>(source_input, aw_bt, o_out);

    // copy-attention: fused sigmoid(enc@copyW^T)·h_new, then softmax over T
    cvt_bf16_kernel<<<(H2 * (H2 / 2) + 255) / 256, 256, 0, stream>>>(copyWb, copy_W, H2 * (H2 / 2));
    ca_caw_kernel<<<BB * TT / 16, 32, 0, stream>>>(enc_out, copyWb, copy_b, h_new, caw_log);
    softmax_rows_kernel<<<BB, 256, 0, stream>>>(caw_log, o_caw, nullptr, TT);
}
